// HM_LSTMCell_13778255086021
// MI455X (gfx1250) — matrix operand order repaired and checked by reading, not compiled, not run on hardware
//
#include <hip/hip_runtime.h>
#include <hip/hip_bf16.h>
#include <math.h>

// ---------------------------------------------------------------------------
// HM-LSTM cell for MI455X (gfx1250, wave32, WMMA).
// f_s = [W_01|U_21|U_11] @ [h_bottom; h_top*z; h*z_bottom] + bias  (K=3072)
// Gates fused with epilogue; bf16 WMMA with f32 accumulation.
// Global->LDS staging uses gfx1250 async-to-LDS with double buffering.
// ---------------------------------------------------------------------------

typedef __attribute__((ext_vector_type(16))) __bf16 v16bf;
typedef __attribute__((ext_vector_type(8)))  float v8f;

static constexpr int Hd  = 1024;   // hidden
static constexpr int Bn  = 4096;   // batch
static constexpr int Kc  = 3072;   // concatenated K
static constexpr int TM  = 32;     // M tile (per gate)
static constexpr int TN  = 128;    // N tile
static constexpr int KK  = 64;     // K step per stage (2 WMMA k-slices)

// round-to-nearest-even f32 -> bf16 bits
__device__ __forceinline__ unsigned short f2bf(float x) {
  unsigned u = __float_as_uint(x);
  u += 0x7fffu + ((u >> 16) & 1u);
  return (unsigned short)(u >> 16);
}
__device__ __forceinline__ float bf2f(unsigned short b) {
  return __uint_as_float(((unsigned)b) << 16);
}
__device__ __forceinline__ float sigmoidf(float x) {
  return 1.0f / (1.0f + expf(-x));
}

// gfx1250 async global->LDS copy, 16B per lane, GVS addressing.
// lds_off: per-lane LDS byte offset (VDST); goff: per-lane byte offset (VADDR);
// base: uniform 64-bit base (SADDR).
__device__ __forceinline__ void async_b128(unsigned lds_off, unsigned goff,
                                           unsigned long long base) {
  asm volatile("global_load_async_to_lds_b128 %0, %1, %2"
               :
               : "v"(lds_off), "v"(goff), "s"(base)
               : "memory");
}
__device__ __forceinline__ void wait_async_le8() {
  asm volatile("s_wait_asynccnt 0x8" ::: "memory");
}
__device__ __forceinline__ void wait_async_0() {
  asm volatile("s_wait_asynccnt 0x0" ::: "memory");
}

// ---------------------------------------------------------------------------
// Kernel 1: pack A_cat bf16 [4097 x 3072] = [W_01 | U_21 | U_11]
// ---------------------------------------------------------------------------
__global__ void pack_A(const float* __restrict__ W01,
                       const float* __restrict__ U21,
                       const float* __restrict__ U11,
                       unsigned short* __restrict__ Acat) {
  const int r = blockIdx.x;              // 0..4096
  const int t = threadIdx.x;             // 256
  unsigned short* dst = Acat + (size_t)r * Kc;
#pragma unroll
  for (int i = 0; i < Kc / 256; ++i) {
    int k = t + i * 256;
    float v;
    if (k < 1024)       v = W01[(size_t)r * 1024 + k];
    else if (k < 2048)  v = U21[(size_t)r * 1024 + (k - 1024)];
    else                v = U11[(size_t)r * 1024 + (k - 2048)];
    dst[k] = f2bf(v);
  }
}

// ---------------------------------------------------------------------------
// Kernel 2: pack B_catT bf16 [4096 x 3072] (N-major, K contiguous)
// 64x64 LDS tile transpose: coalesced f32 reads, 16B bf16 writes.
// ---------------------------------------------------------------------------
__global__ void pack_B(const float* __restrict__ h_bottom,
                       const float* __restrict__ h_top,
                       const float* __restrict__ h,
                       const float* __restrict__ z,
                       const float* __restrict__ zb,
                       unsigned short* __restrict__ BcatT) {
  __shared__ __align__(16) unsigned short tile[64][72];
  const int t      = threadIdx.x;          // 256
  const int n_base = blockIdx.x * 64;
  const int k_base = blockIdx.y * 64;      // segment-uniform

  const float* src;
  int seg;
  if (k_base < 1024)      { src = h_bottom; seg = 0; }
  else if (k_base < 2048) { src = h_top;    seg = 1; }
  else                    { src = h;        seg = 2; }
  const int k_src = k_base - seg * 1024;

#pragma unroll
  for (int i = 0; i < 16; ++i) {
    int e     = t + i * 256;
    int k_loc = e >> 6;
    int n_loc = e & 63;
    int n     = n_base + n_loc;
    float s   = (seg == 0) ? 1.0f : (seg == 1 ? z[n] : zb[n]);
    float v   = src[(size_t)(k_src + k_loc) * Bn + n] * s;
    tile[n_loc][k_loc] = f2bf(v);
  }
  __syncthreads();

#pragma unroll
  for (int i = 0; i < 2; ++i) {
    int c     = t + i * 256;             // 0..511
    int n_loc = c >> 3;
    int kc    = c & 7;
    uint4 d   = *(const uint4*)&tile[n_loc][kc * 8];
    *(uint4*)(BcatT + (size_t)(n_base + n_loc) * Kc + k_base + kc * 8) = d;
  }
}

// ---------------------------------------------------------------------------
// Kernel 3: fused 4-gate GEMM + LSTM epilogue.
// Block: 256 thr (8 waves). Tile: TM=32 rows (x4 gates) x TN=128 cols.
// Wave w owns columns [w*16, w*16+16); computes 4 gates x 2 M-frags.
// Double-buffered async global->LDS staging (ASYNCcnt pipeline).
// ---------------------------------------------------------------------------
__global__ __launch_bounds__(256)
void hmlstm_gemm(const unsigned short* __restrict__ Acat,
                 const unsigned short* __restrict__ BcatT,
                 const float* __restrict__ bias,
                 const float* __restrict__ Cold,
                 const float* __restrict__ Hold,
                 const float* __restrict__ Zv,
                 const float* __restrict__ Zb,
                 float* __restrict__ out) {
  __shared__ __align__(16) unsigned short sA[2][128][72];  // 4 gates x 32 rows
  __shared__ __align__(16) unsigned short sB[2][128][72];  // 128 cols

  const int t      = threadIdx.x;
  const int lane   = t & 31;
  const int w      = t >> 5;             // wave id 0..7
  const int m_base = blockIdx.y * TM;    // 0..992
  const int n_base = blockIdx.x * TN;    // 0..3968
  const int NT     = Kc / KK;            // 48 K-stages

  const unsigned long long baseA = (unsigned long long)(uintptr_t)Acat;
  const unsigned long long baseB = (unsigned long long)(uintptr_t)BcatT;

  // precompute this thread's 4 staging chunks (row, kc) and LDS/global offsets
  // chunk c = t + i*256 : row = c>>3 (0..127), kc = c&7 (16B chunks along K)
  unsigned ldsA[4], ldsB[4], gArow[4], gBrow[4];
#pragma unroll
  for (int i = 0; i < 4; ++i) {
    int c   = t + i * 256;
    int row = c >> 3;
    int kc  = c & 7;
    int g   = row >> 5;
    int m   = row & 31;
    ldsA[i]  = (unsigned)(uintptr_t)&sA[0][row][kc * 8];
    ldsB[i]  = (unsigned)(uintptr_t)&sB[0][row][kc * 8];
    gArow[i] = (unsigned)(((size_t)(g * Hd + m_base + m) * Kc + kc * 8) * 2);
    gBrow[i] = (unsigned)(((size_t)(n_base + row) * Kc + kc * 8) * 2);
  }
  const unsigned bufStride = (unsigned)(sizeof(unsigned short) * 128 * 72);

  auto issue_stage = [&](int buf, int kt) {
    const unsigned kb = (unsigned)(kt * KK * 2);   // byte offset along K
    const unsigned bo = buf ? bufStride : 0u;
#pragma unroll
    for (int i = 0; i < 4; ++i) async_b128(ldsA[i] + bo, gArow[i] + kb, baseA);
#pragma unroll
    for (int i = 0; i < 4; ++i) async_b128(ldsB[i] + bo, gBrow[i] + kb, baseB);
  };

  union Frag { uint4 u[2]; v16bf v; };

  v8f acc[4][2];
  const v8f zero8 = {0.f, 0.f, 0.f, 0.f, 0.f, 0.f, 0.f, 0.f};
#pragma unroll
  for (int g = 0; g < 4; ++g)
#pragma unroll
    for (int mi = 0; mi < 2; ++mi) acc[g][mi] = zero8;

  issue_stage(0, 0);                      // prologue: stage 0 in flight

  for (int kt = 0; kt < NT; ++kt) {
    const int buf = kt & 1;
    if (kt + 1 < NT) {
      issue_stage(buf ^ 1, kt + 1);       // prefetch next K-slab
      wait_async_le8();                   // loads return in order: stage kt done
    } else {
      wait_async_0();
    }
    __syncthreads();                      // all waves' stage-kt writes visible

#pragma unroll
    for (int ks = 0; ks < KK; ks += 32) {
      // B fragment: lanes 16-31 hold K+8 / K+24 halves (matches A)
      const int kB  = ks + ((lane >> 4) << 3);   // +0 / +8
      const int col = (w << 4) + (lane & 15);
      Frag bf;
      bf.u[0] = *(const uint4*)&sB[buf][col][kB];
      bf.u[1] = *(const uint4*)&sB[buf][col][kB + 16];
      // A fragments: lanes 16-31 hold K+8 / K+24 halves
      const int kA = ks + ((lane >> 4) << 3);    // +0 / +8
#pragma unroll
      for (int g = 0; g < 4; ++g) {
#pragma unroll
        for (int mi = 0; mi < 2; ++mi) {
          const int row = (g << 5) + (mi << 4) + (lane & 15);
          Frag af;
          af.u[0] = *(const uint4*)&sA[buf][row][kA];
          af.u[1] = *(const uint4*)&sA[buf][row][kA + 16];
          acc[g][mi] = __builtin_amdgcn_wmma_f32_16x16x32_bf16(
              false, af.v, false, bf.v, (short)0, acc[g][mi], false, false);
        }
      }
    }
    __syncthreads();   // everyone done reading buf before stage kt+2 overwrites
  }

  // ---- fused epilogue: bias + gates + c_new / h_new ----
  const int n  = n_base + (w << 4) + (lane & 15);
  const float z   = Zv[n];
  const float zb  = Zb[n];
  const float omz = 1.0f - z;
  float* out_h = out;
  float* out_c = out + (size_t)Hd * Bn;

#pragma unroll
  for (int mi = 0; mi < 2; ++mi) {
    const int mrow0 = m_base + (mi << 4) + ((lane >> 4) << 3);  // +0 / +8
#pragma unroll
    for (int v = 0; v < 8; ++v) {
      const int m = mrow0 + v;
      float fs_f = acc[0][mi][v] + bias[m];
      float fs_i = acc[1][mi][v] + bias[m + Hd];
      float fs_o = acc[2][mi][v] + bias[m + 2 * Hd];
      float fs_g = acc[3][mi][v] + bias[m + 3 * Hd];
      float fg = sigmoidf(fs_f);
      float ig = sigmoidf(fs_i);
      float og = sigmoidf(fs_o);
      float gg = tanhf(fs_g);
      size_t idx = (size_t)m * Bn + n;
      float c_old = Cold[idx];
      float h_old = Hold[idx];
      float iggg  = ig * gg;
      float c_new = z * iggg + omz * (1.0f - zb) * c_old
                  + omz * zb * (fg * c_old + iggg);
      float tc    = tanhf(c_new);
      float h_new = z * og * tc + omz * (1.0f - zb) * h_old
                  + omz * zb * og * tc;
      out_h[idx] = h_new;
      out_c[idx] = c_new;
    }
  }
}

// ---------------------------------------------------------------------------
// Kernel 4: boundary row (row 4H) GEMV + straight-through z_new.
// ---------------------------------------------------------------------------
__global__ void zrow_kernel(const unsigned short* __restrict__ Acat,
                            const unsigned short* __restrict__ BcatT,
                            const float* __restrict__ bias,
                            float* __restrict__ out_z) {
  __shared__ unsigned short sW[Kc];     // A_cat row 4H (6 KB)
  const int t    = threadIdx.x;
  const int lane = t & 31;
  const int w    = t >> 5;
  const unsigned short* wrow = Acat + (size_t)(4 * Hd) * Kc;
#pragma unroll
  for (int i = 0; i < Kc / 256; ++i) sW[t + i * 256] = wrow[t + i * 256];
  __syncthreads();

  const int n = blockIdx.x * 8 + w;
  const unsigned short* brow = BcatT + (size_t)n * Kc;
  float acc = 0.0f;
  for (int chunk = lane; chunk < Kc / 8; chunk += 32) {
    uint4 bq = *(const uint4*)(brow + chunk * 8);
    const unsigned* bu = (const unsigned*)&bq;
#pragma unroll
    for (int j = 0; j < 4; ++j) {
      unsigned u = bu[j];
      int k = chunk * 8 + j * 2;
      acc += bf2f((unsigned short)(u & 0xffffu)) * bf2f(sW[k]);
      acc += bf2f((unsigned short)(u >> 16))     * bf2f(sW[k + 1]);
    }
  }
#pragma unroll
  for (int off = 16; off > 0; off >>= 1) acc += __shfl_xor(acc, off, 32);
  if (lane == 0) {
    float s  = acc + bias[4 * Hd];
    float zh = fminf(fmaxf((s * 1.0f + 1.0f) * 0.5f, 0.0f), 1.0f);  // A = 1
    out_z[n] = (zh > 0.5f) ? 1.0f : 0.0f;   // straight-through forward value
  }
}

// ---------------------------------------------------------------------------
extern "C" void kernel_launch(void* const* d_in, const int* in_sizes, int n_in,
                              void* d_out, int out_size, void* d_ws, size_t ws_size,
                              hipStream_t stream) {
  const float* c        = (const float*)d_in[0];
  const float* h_bottom = (const float*)d_in[1];
  const float* h        = (const float*)d_in[2];
  const float* h_top    = (const float*)d_in[3];
  const float* z        = (const float*)d_in[4];
  const float* z_bottom = (const float*)d_in[5];
  const float* U_11     = (const float*)d_in[6];
  const float* U_21     = (const float*)d_in[7];
  const float* W_01     = (const float*)d_in[8];
  const float* bias     = (const float*)d_in[9];
  float* out = (float*)d_out;

  // workspace: A_cat bf16 [4097 x 3072] then B_catT bf16 [4096 x 3072]
  unsigned short* Acat  = (unsigned short*)d_ws;
  unsigned short* BcatT = Acat + (size_t)(4 * Hd + 1) * Kc;

  pack_A<<<4 * Hd + 1, 256, 0, stream>>>(W_01, U_21, U_11, Acat);
  pack_B<<<dim3(Bn / 64, Kc / 64), 256, 0, stream>>>(
      h_bottom, h_top, h, z, z_bottom, BcatT);
  hmlstm_gemm<<<dim3(Bn / TN, Hd / TM), 256, 0, stream>>>(
      Acat, BcatT, bias, c, h, z, z_bottom, out);
  zrow_kernel<<<Bn / 8, 256, 0, stream>>>(
      Acat, BcatT, bias, out + (size_t)2 * Hd * Bn);
}
